// MambaBlock_90829968376157
// MI455X (gfx1250) — compile-verified, matrix operand order checked by reading, not run on hardware
//
#include <hip/hip_runtime.h>

// ---------------------------------------------------------------------------
// Types / WMMA helpers
// ---------------------------------------------------------------------------
typedef __attribute__((ext_vector_type(16))) __bf16 v16bf;
typedef __attribute__((ext_vector_type(8)))  float  v8f;

#define MBLK 128
#define NBLK 128
#define KSTEP 32
#define LDSS 40   // padded LDS row stride in bf16 elements (bank-conflict pad)

// --------- CDNA5 async global->LDS copy (ASYNCcnt path), guarded -----------
#if __has_builtin(__builtin_amdgcn_global_load_async_to_lds_b128) && \
    __has_builtin(__builtin_amdgcn_s_wait_asynccnt)
#define USE_ASYNC_LDS 1
#else
#define USE_ASYNC_LDS 0
#endif

// builtin prototype (from compiler diagnostic): (v4i AS1*, v4i AS3*, imm, imm)
typedef int v4i_gcc __attribute__((vector_size(16)));
typedef __attribute__((address_space(1))) v4i_gcc* gv4p;   // global int4*
typedef __attribute__((address_space(3))) v4i_gcc* lv4p;   // LDS int4*

// copy 16 bytes per lane, global -> LDS, no VGPR round-trip (ASYNCcnt)
__device__ __forceinline__ void async_cp16(const void* g, const void* l) {
#if USE_ASYNC_LDS
  __builtin_amdgcn_global_load_async_to_lds_b128(
      (gv4p)(unsigned long long)g,                    // generic global == AS1 value
      (lv4p)(unsigned)(unsigned long long)l,          // low 32 bits = LDS offset
      0, 0);
#else
  (void)g; (void)l;
#endif
}
__device__ __forceinline__ void async_wait0() {
#if USE_ASYNC_LDS
  __builtin_amdgcn_s_wait_asynccnt(0);
#endif
}

union Frag { v16bf v; unsigned u[8]; };

// Gather one 16x32 (A) or 16(N)x32(K) (B, stored [n][k]) fragment from LDS
// following the CDNA5 16-bit A-matrix VGPR layout:
//   lane l: row = base + (l&15), khalf = l>>4
//   vgpr v: k = (v>>2)*16 + khalf*8 + (v&3)*2 (+0/1 packed in one dword)
__device__ __forceinline__ void load_frag(const __bf16* smem, int rowloc, int lhalf, Frag& f) {
#pragma unroll
  for (int v = 0; v < 8; ++v) {
    const int kbase = ((v >> 2) << 4) + (lhalf << 3) + ((v & 3) << 1);
    f.u[v] = *reinterpret_cast<const unsigned*>(smem + rowloc * LDSS + kbase);
  }
}

// ---------------------------------------------------------------------------
// Generic bf16 WMMA GEMM:  C(MxN, f32, ldc) = A(MxK, bf16, lda) * B(KxN, bf16, ldb)
// M multiple of 128, K multiple of 32, N multiple of 16 (guarded vs NBLK).
// ---------------------------------------------------------------------------
__global__ __launch_bounds__(256)
void gemm_bf16_kernel(const __bf16* __restrict__ A, const __bf16* __restrict__ B,
                      float* __restrict__ C, int M, int N, int K,
                      int lda, int ldb, int ldc) {
  __shared__ __align__(16) __bf16 As[MBLK * LDSS];   // [m][k]
  __shared__ __align__(16) __bf16 Bs[NBLK * LDSS];   // [n][k] (transposed)
  const int tid   = threadIdx.x;
  const int m0    = blockIdx.y * MBLK;
  const int n0    = blockIdx.x * NBLK;
  const int wave  = tid >> 5;
  const int lane  = tid & 31;
  const int wm    = (wave >> 1) << 5;   // 0,32,64,96
  const int wn    = (wave & 1) << 6;    // 0,64
  const int lhalf = lane >> 4;
  const int lmod  = lane & 15;

  v8f acc[2][4] = {};

  const int ra = tid >> 1;            // A tile row 0..127
  const int ha = (tid & 1) << 4;      // 0 / 16 (k-half)
  const int kb = tid >> 3;            // B tile k-row 0..31
  const int nb = (tid & 7) << 4;      // 0..112 (n segment)

  for (int k0 = 0; k0 < K; k0 += KSTEP) {
    __syncthreads();
#if USE_ASYNC_LDS
    {   // A tile: 32 bytes/thread straight copy, ASYNCcnt DMA path
      const __bf16* gsrc = A + (size_t)(m0 + ra) * lda + k0 + ha;
      __bf16* ldst = &As[ra * LDSS + ha];
      async_cp16(gsrc, ldst);
      async_cp16(gsrc + 8, ldst + 8);
    }
#else
    {   // A tile: 32 bytes/thread, straight copy via VGPRs
      const uint4* p = reinterpret_cast<const uint4*>(A + (size_t)(m0 + ra) * lda + k0 + ha);
      uint4 q0 = p[0], q1 = p[1];
      uint4* sp = reinterpret_cast<uint4*>(&As[ra * LDSS + ha]);
      sp[0] = q0; sp[1] = q1;
    }
#endif
    {   // B tile: load 16 n-contiguous bf16, transpose-store to [n][k]
      uint4 q0 = {0, 0, 0, 0}, q1 = {0, 0, 0, 0};
      const int nn = n0 + nb;
      if (nn < N) {
        const uint4* p = reinterpret_cast<const uint4*>(B + (size_t)(k0 + kb) * ldb + nn);
        q0 = p[0]; q1 = p[1];
      }
      __bf16 tmp[16];
      *reinterpret_cast<uint4*>(&tmp[0]) = q0;
      *reinterpret_cast<uint4*>(&tmp[8]) = q1;
#pragma unroll
      for (int e = 0; e < 16; ++e) Bs[(nb + e) * LDSS + kb] = tmp[e];
    }
    if (k0 + KSTEP < K) {   // global_prefetch_b8 of next tiles
      __builtin_prefetch(A + (size_t)(m0 + ra) * lda + k0 + KSTEP + ha, 0, 1);
      __builtin_prefetch(B + (size_t)(k0 + KSTEP + kb) * ldb + n0 + nb, 0, 1);
    }
    async_wait0();
    __syncthreads();

    Frag af[2], bfr[4];
#pragma unroll
    for (int mt = 0; mt < 2; ++mt) load_frag(As, wm + (mt << 4) + lmod, lhalf, af[mt]);
#pragma unroll
    for (int nt = 0; nt < 4; ++nt) load_frag(Bs, wn + (nt << 4) + lmod, lhalf, bfr[nt]);
#pragma unroll
    for (int mt = 0; mt < 2; ++mt)
#pragma unroll
      for (int nt = 0; nt < 4; ++nt)
        acc[mt][nt] = __builtin_amdgcn_wmma_f32_16x16x32_bf16(
            false, af[mt].v, false, bfr[nt].v, (short)0, acc[mt][nt], false, false);
  }

  // Epilogue: C/D layout — vgpr r: row = base + lhalf*8 + r, col = base + lmod
#pragma unroll
  for (int mt = 0; mt < 2; ++mt)
#pragma unroll
    for (int nt = 0; nt < 4; ++nt) {
      const int n = n0 + wn + (nt << 4) + lmod;
      if (n < N) {
#pragma unroll
        for (int r = 0; r < 8; ++r) {
          const int m = m0 + wm + (mt << 4) + (lhalf << 3) + r;
          C[(size_t)m * ldc + n] = acc[mt][nt][r];
        }
      }
    }
}

// ---------------------------------------------------------------------------
// Causal full conv (K=4 taps) as 4 accumulated implicit GEMMs + bias + SiLU.
// U: (4096 x 2048) bf16 rows g=b*2048+t;  W: (4,2048,2048) bf16;  out bf16.
// out[t] = silu( sum_w U[t+w-3] @ W[w] + bias )  with zero pad per sequence.
// ---------------------------------------------------------------------------
__global__ __launch_bounds__(256)
void conv_silu_kernel(const __bf16* __restrict__ U, const __bf16* __restrict__ W,
                      const float* __restrict__ bias, __bf16* __restrict__ out) {
  constexpr int LDA = 2048, LDB = 2048, LDC = 2048, KDIM = 2048;
  __shared__ __align__(16) __bf16 As[MBLK * LDSS];
  __shared__ __align__(16) __bf16 Bs[NBLK * LDSS];
  const int tid   = threadIdx.x;
  const int m0    = blockIdx.y * MBLK;
  const int n0    = blockIdx.x * NBLK;
  const int wave  = tid >> 5;
  const int lane  = tid & 31;
  const int wm    = (wave >> 1) << 5;
  const int wn    = (wave & 1) << 6;
  const int lhalf = lane >> 4;
  const int lmod  = lane & 15;

  v8f acc[2][4] = {};

  const int ra = tid >> 1;
  const int ha = (tid & 1) << 4;
  const int kb = tid >> 3;
  const int nb = (tid & 7) << 4;
  const int g    = m0 + ra;      // global row (b*2048 + t); 128 | 2048 so block is 1 batch
  const int tpos = g & 2047;

  for (int w = 0; w < 4; ++w) {
    const __bf16* Bw   = W + (size_t)w * KDIM * LDB;
    const bool   valid = (tpos + w) >= 3;                  // causal zero-pad
    const __bf16* Arow = U + (size_t)(g + w - 3) * LDA;
    for (int k0 = 0; k0 < KDIM; k0 += KSTEP) {
      __syncthreads();
#if USE_ASYNC_LDS
      if (valid) {
        const __bf16* gsrc = Arow + k0 + ha;
        __bf16* ldst = &As[ra * LDSS + ha];
        async_cp16(gsrc, ldst);
        async_cp16(gsrc + 8, ldst + 8);
      } else {
        const uint4 z = {0, 0, 0, 0};
        uint4* sp = reinterpret_cast<uint4*>(&As[ra * LDSS + ha]);
        sp[0] = z; sp[1] = z;
      }
#else
      {
        uint4 q0 = {0, 0, 0, 0}, q1 = {0, 0, 0, 0};
        if (valid) {
          const uint4* p = reinterpret_cast<const uint4*>(Arow + k0 + ha);
          q0 = p[0]; q1 = p[1];
        }
        uint4* sp = reinterpret_cast<uint4*>(&As[ra * LDSS + ha]);
        sp[0] = q0; sp[1] = q1;
      }
#endif
      {
        const uint4* p = reinterpret_cast<const uint4*>(Bw + (size_t)(k0 + kb) * LDB + n0 + nb);
        uint4 q0 = p[0], q1 = p[1];
        __bf16 tmp[16];
        *reinterpret_cast<uint4*>(&tmp[0]) = q0;
        *reinterpret_cast<uint4*>(&tmp[8]) = q1;
#pragma unroll
        for (int e = 0; e < 16; ++e) Bs[(nb + e) * LDSS + kb] = tmp[e];
      }
      async_wait0();
      __syncthreads();

      Frag af[2], bfr[4];
#pragma unroll
      for (int mt = 0; mt < 2; ++mt) load_frag(As, wm + (mt << 4) + lmod, lhalf, af[mt]);
#pragma unroll
      for (int nt = 0; nt < 4; ++nt) load_frag(Bs, wn + (nt << 4) + lmod, lhalf, bfr[nt]);
#pragma unroll
      for (int mt = 0; mt < 2; ++mt)
#pragma unroll
        for (int nt = 0; nt < 4; ++nt)
          acc[mt][nt] = __builtin_amdgcn_wmma_f32_16x16x32_bf16(
              false, af[mt].v, false, bfr[nt].v, (short)0, acc[mt][nt], false, false);
    }
  }

  // fused bias + SiLU, store bf16 activations for the SSM stages
#pragma unroll
  for (int mt = 0; mt < 2; ++mt)
#pragma unroll
    for (int nt = 0; nt < 4; ++nt) {
      const int n  = n0 + wn + (nt << 4) + lmod;
      const float bv = bias[n];
#pragma unroll
      for (int r = 0; r < 8; ++r) {
        const int m = m0 + wm + (mt << 4) + (lhalf << 3) + r;
        const float v = acc[mt][nt][r] + bv;
        out[(size_t)m * LDC + n] = (__bf16)(v / (1.f + __expf(-v)));
      }
    }
}

// ---------------------------------------------------------------------------
// Selective scan: one channel per lane, s[16] + A[16] in VGPRs.
// delta: (4096,2048) f32 | u: (4096,2048) bf16 | xdbl: (4096,96) f32 (B=64..79, C=80..95)
// y: (4096,2048) f32 | resid: (4096) f32 = D[d] * sum_n s_final
// grid: 16 blocks x 256 threads  (b = blk>>3, d = (blk&7)*256 + tid)
// ---------------------------------------------------------------------------
__global__ __launch_bounds__(256)
void scan_kernel(const float* __restrict__ delta, const __bf16* __restrict__ u,
                 const float* __restrict__ xdbl, const float* __restrict__ A_log,
                 const float* __restrict__ Dp, float* __restrict__ y,
                 float* __restrict__ resid) {
  const int b   = blockIdx.x >> 3;
  const int d   = ((blockIdx.x & 7) << 8) + threadIdx.x;
  const int tid = threadIdx.x;
  __shared__ __align__(16) float sBC[32][32];  // [t_local][0..15 = B, 16..31 = C]

  float A[16];
#pragma unroll
  for (int n = 0; n < 16; ++n) A[n] = -__expf(A_log[d * 16 + n]);
  float s[16];
#pragma unroll
  for (int n = 0; n < 16; ++n) s[n] = 0.f;

  const size_t gbase = (size_t)b * 2048;
  for (int t0 = 0; t0 < 2048; t0 += 32) {
    __syncthreads();
#if USE_ASYNC_LDS
    {   // stage 32 steps of (B,C): 16 bytes/thread async to LDS
      const int rr = tid >> 3, cc = (tid & 7) << 2;
      async_cp16(&xdbl[(gbase + t0 + rr) * 96 + 64 + cc], &sBC[rr][cc]);
      async_wait0();
    }
#else
#pragma unroll
    for (int j = 0; j < 4; ++j) {
      const int idx = tid * 4 + j;              // 0..1023
      const int rr = idx >> 5, cc = idx & 31;
      sBC[rr][cc] = xdbl[(gbase + t0 + rr) * 96 + 64 + cc];
    }
#endif
    __syncthreads();
    for (int tl = 0; tl < 32; ++tl) {
      const size_t gg = gbase + t0 + tl;
      const float dv = delta[gg * 2048 + d];
      const float uv = (float)u[gg * 2048 + d];
      const float du = dv * uv;
      float accy = 0.f;
#pragma unroll
      for (int n = 0; n < 16; ++n) {
        const float dA = __expf(dv * A[n]);
        s[n] = dA * s[n] + du * sBC[tl][n];
        accy += s[n] * sBC[tl][16 + n];
      }
      y[gg * 2048 + d] = accy;
    }
  }
  float r = 0.f;
#pragma unroll
  for (int n = 0; n < 16; ++n) r += s[n];
  resid[gbase + d] = r * Dp[d];
}

// ---------------------------------------------------------------------------
// Elementwise helpers
// ---------------------------------------------------------------------------
__global__ void cvt2d_f32_bf16(const float* __restrict__ src, __bf16* __restrict__ dst,
                               long long rows, long long cols, int lds, int ldd) {
  const long long total = rows * cols;
  for (long long i = (long long)blockIdx.x * blockDim.x + threadIdx.x; i < total;
       i += (long long)gridDim.x * blockDim.x) {
    const long long r = i / cols, c = i % cols;
    dst[r * ldd + c] = (__bf16)src[r * lds + c];
  }
}

__global__ void softplus_bias_kernel(float* __restrict__ x, const float* __restrict__ bias,
                                     int cols, long long total) {
  for (long long i = (long long)blockIdx.x * blockDim.x + threadIdx.x; i < total;
       i += (long long)gridDim.x * blockDim.x) {
    const float v = x[i] + bias[i % cols];
    x[i] = (v > 20.f) ? v : log1pf(__expf(v));
  }
}

// ybf = bf16( (y + resid[b,d]) * silu(res) )
__global__ void final_ew_kernel(const float* __restrict__ y, const float* __restrict__ resid,
                                const float* __restrict__ res, __bf16* __restrict__ out,
                                long long total) {
  for (long long i = (long long)blockIdx.x * blockDim.x + threadIdx.x; i < total;
       i += (long long)gridDim.x * blockDim.x) {
    const long long d = i & 2047;
    const long long g = i >> 11;
    const long long b = g >> 11;
    const float rv = resid[(b << 11) + d];
    const float rr = res[i];
    const float sl = rr / (1.f + __expf(-rr));
    out[i] = (__bf16)((y[i] + rv) * sl);
  }
}

// ---------------------------------------------------------------------------
// Host orchestration
// ---------------------------------------------------------------------------
extern "C" void kernel_launch(void* const* d_in, const int* in_sizes, int n_in,
                              void* d_out, int out_size, void* d_ws, size_t ws_size,
                              hipStream_t stream) {
  (void)in_sizes; (void)n_in; (void)out_size; (void)ws_size;
  const float* x      = (const float*)d_in[0];   // (2,2048,1024)
  const float* W_in   = (const float*)d_in[1];   // (1024,4096)
  const float* conv_w = (const float*)d_in[2];   // (4,2048,2048)
  const float* conv_b = (const float*)d_in[3];   // (2048)
  const float* W_x    = (const float*)d_in[4];   // (2048,96)
  const float* W_dt   = (const float*)d_in[5];   // (64,2048)
  const float* b_dt   = (const float*)d_in[6];   // (2048)
  const float* A_log  = (const float*)d_in[7];   // (2048,16)
  const float* Dp     = (const float*)d_in[8];   // (2048)
  const float* W_out  = (const float*)d_in[9];   // (2048,1024)
  float* out = (float*)d_out;                    // (2,2048,1024)

  const long long BL = 4096;                     // B*L
  char* ws = (char*)d_ws;
  size_t off = 0;
  auto alloc = [&](size_t bytes) { size_t r = off; off += (bytes + 255) & ~(size_t)255; return r; };
  __bf16* XBF    = (__bf16*)(ws + alloc((size_t)BL * 1024 * 2));      // x bf16
  __bf16* WARENA = (__bf16*)(ws + alloc((size_t)4 * 2048 * 2048 * 2)); // per-stage bf16 weights
  float*  UF32   = (float*) (ws + alloc((size_t)BL * 2048 * 4));      // u_raw f32 -> delta f32
  float*  RESB   = (float*) (ws + alloc((size_t)BL * 2048 * 4));      // res branch f32
  __bf16* UBF    = (__bf16*)(ws + alloc((size_t)BL * 2048 * 2));      // u_raw bf16 -> y bf16
  __bf16* UACT   = (__bf16*)(ws + alloc((size_t)BL * 2048 * 2));      // silu(conv) bf16
  float*  XDBL   = (float*) (ws + alloc((size_t)BL * 96 * 4));        // x_dbl f32
  __bf16* DXBF   = (__bf16*)(ws + alloc((size_t)BL * 64 * 2));        // delta-rank bf16
  float*  YF32   = (float*) (ws + alloc((size_t)BL * 2048 * 4));      // scan output f32
  float*  RESID  = (float*) (ws + alloc((size_t)BL * 4));             // final-state residual

  const dim3 blk(256);
  auto cvt = [&](const float* s, __bf16* d, long long rows, long long cols, int lds, int ldd) {
    long long tot = rows * cols;
    long long nb = (tot + 255) / 256; if (nb > 4096) nb = 4096;
    cvt2d_f32_bf16<<<dim3((unsigned)nb), blk, 0, stream>>>(s, d, rows, cols, lds, ldd);
  };
  auto gemm = [&](const __bf16* A, int lda, const __bf16* B, int ldb,
                  float* C, int ldc, int M, int N, int K) {
    dim3 grid((N + NBLK - 1) / NBLK, M / MBLK);
    gemm_bf16_kernel<<<grid, blk, 0, stream>>>(A, B, C, M, N, K, lda, ldb, ldc);
  };

  // 1) in-projection: split W_in into u / res halves (column offset into same B)
  cvt(x, XBF, BL, 1024, 1024, 1024);
  cvt(W_in, WARENA, 1024, 4096, 4096, 4096);
  gemm(XBF, 1024, WARENA,        4096, UF32, 2048, 4096, 2048, 1024);  // u_raw
  gemm(XBF, 1024, WARENA + 2048, 4096, RESB, 2048, 4096, 2048, 1024);  // res

  // 2) causal full conv (4 taps) + bias + SiLU, fused into WMMA epilogue
  cvt(UF32, UBF, BL, 2048, 2048, 2048);
  cvt(conv_w, WARENA, (long long)4 * 2048, 2048, 2048, 2048);
  conv_silu_kernel<<<dim3(2048 / NBLK, 4096 / MBLK), blk, 0, stream>>>(UBF, WARENA, conv_b, UACT);

  // 3) x_dbl = u @ W_x   (N=96: guarded partial tile)
  cvt(W_x, WARENA, 2048, 96, 96, 96);
  gemm(UACT, 2048, WARENA, 96, XDBL, 96, 4096, 96, 2048);

  // 4) delta = softplus(x_dbl[:, :64] @ W_dt + b_dt)   (reuse UF32 as delta)
  cvt(XDBL, DXBF, BL, 64, 96, 64);
  cvt(W_dt, WARENA, 64, 2048, 2048, 2048);
  gemm(DXBF, 64, WARENA, 2048, UF32, 2048, 4096, 2048, 64);
  softplus_bias_kernel<<<dim3(4096), blk, 0, stream>>>(UF32, b_dt, 2048, BL * 2048);

  // 5) selective scan (sequential in L, parallel over 4096 channels)
  scan_kernel<<<dim3(16), blk, 0, stream>>>(UF32, UACT, XDBL, A_log, Dp, YF32, RESID);

  // 6) gating + residual, convert to bf16 (reuse UBF)
  final_ew_kernel<<<dim3(4096), blk, 0, stream>>>(YF32, RESID, RESB, UBF, BL * 2048);

  // 7) out-projection
  cvt(W_out, WARENA, 2048, 1024, 1024, 1024);
  gemm(UBF, 2048, WARENA, 1024, out, 1024, 4096, 1024, 2048);
}
